// DiscMakerDNC_53274774339936
// MI455X (gfx1250) — compile-verified, matrix operand order checked by reading, not run on hardware
//
#include <hip/hip_runtime.h>
#include <hip/hip_bf16.h>

typedef _Float16 v16h __attribute__((ext_vector_type(16)));
typedef _Float16 v8h  __attribute__((ext_vector_type(8)));
typedef float    v8f  __attribute__((ext_vector_type(8)));

#define NB 32
#define NL 2048
#define NS 32
#define NC 64
#define NTC 17
#define NH 32
#define GAMMA_C 0.5f
#define CH 32              /* x/y staging chunk (steps) */

#if __has_builtin(__builtin_amdgcn_rcpf)
  #define RCPF(x) __builtin_amdgcn_rcpf(x)
#else
  #define RCPF(x) (1.0f / (x))
#endif

__device__ __forceinline__ float fast_tanh(float x) {
    // 1 - 2/(e^{2x}+1): v_exp + v_rcp, saturates correctly at +-1
    float e = __expf(2.0f * x);
    return 1.0f - 2.0f * RCPF(e + 1.0f);
}

#if __has_builtin(__builtin_amdgcn_tanhf)
  #define TANHF(x) __builtin_amdgcn_tanhf(x)
#elif __has_builtin(__builtin_amdgcn_tanh_f32)
  #define TANHF(x) __builtin_amdgcn_tanh_f32(x)
#else
  #define TANHF(x) fast_tanh(x)
#endif

#if __has_builtin(__builtin_amdgcn_global_load_async_to_lds_b32) && __has_builtin(__builtin_amdgcn_s_wait_asynccnt)
  #define ASYNC_CP_B32(gsrc, ldst)                                             \
      __builtin_amdgcn_global_load_async_to_lds_b32(                           \
          (__attribute__((address_space(1))) int*)(gsrc),                      \
          (__attribute__((address_space(3))) int*)(ldst), 0, 0)
  #define ASYNC_WAIT() __builtin_amdgcn_s_wait_asynccnt(0)
#else
  #define ASYNC_CP_B32(gsrc, ldst) (*(float*)(ldst) = *(const float*)(gsrc))
  #define ASYNC_WAIT() ((void)0)
#endif

// ---------------- LDS layout (bytes) ----------------
#define OFF_WCT    0        /* 131072: Wc transposed [c][d][s], f16 */
#define OFF_WXT1   131072   /* 8192 */
#define OFF_WHT1   139264   /* 8192 */
#define OFF_WXT2   147456   /* 8192 */
#define OFF_WHT2   155648   /* 8192 */
#define OFF_WOT    163840   /* 2048 */
#define OFF_WLT    165888   /* 2048 */
#define OFF_WD1T   167936   /* 4096 */
#define OFF_WD2T   172032   /* 8192 */
#define OFF_INP16  180224   /* 2048 */
#define OFF_H1_16  182272   /* 2048 */
#define OFF_H2_16  184320   /* 2048 */
#define OFF_ST16   186368   /* 2048 */
#define OFF_TMPO16 188416   /* 2048 */
#define OFF_CO16   190464   /* 2048 */
#define OFF_HID16  192512   /* 4096 */
#define OFF_STATE  196608   /* 4096 */
#define OFF_STNEW  200704   /* 4096 */
#define OFF_Z      204800   /* 16384: shared by LSTM1/LSTM2 (disjoint phases) */
#define OFF_C1F    221184   /* 4096 */
#define OFF_C2F    225280   /* 4096 */
#define OFF_INP32  229376   /* 4096 */
#define OFF_GATEA  233472   /* 8192 */
#define OFF_GATEB  241664   /* 8192 */
#define OFF_GSM    249856   /* 8192 */
#define OFF_CO32   258048   /* 4096 */
#define OFF_UXL    262144   /* 8192 */
#define OFF_BCL    270336   /* 8192 */
#define OFF_MUL    278528   /* 2304 */
#define OFF_B1L    280832   /* 512 */
#define OFF_B2L    281344   /* 512 */
#define OFF_BOL    281856   /* 128 */
#define OFF_BLINL  281984   /* 128 */
#define OFF_BD1L   282112   /* 256 */
#define OFF_BD2L   282368   /* 256 */
#define OFF_XS     282624   /* 128 */
#define OFF_ERR    282752   /* 128 */
#define OFF_TH     282880   /* 128 */
#define OFF_XT     283008   /* 8192: x staging, 2 x (32b x 32t) f32 */
#define OFF_YT     291200   /* 8192: y staging */
#define SMEM_TOTAL 299392

__device__ __forceinline__ float sigm(float x) { return 0.5f + 0.5f * TANHF(0.5f * x); }

// f16 fragment load from LDS (row-major, `stride` halfs per row).
__device__ __forceinline__ v16h ldsFrag16s(const _Float16* base, int rowBase, int stride, int lane) {
    int m  = lane & 15;
    int hi = (lane >> 4) & 1;
    const _Float16* p = base + (rowBase + m) * stride + hi * 8;
    union { v16h v; v8h h[2]; } u;
    u.h[0] = *(const v8h*)(p);
    u.h[1] = *(const v8h*)(p + 16);
    return u.v;
}

#define WMMA16(A, B, C) __builtin_amdgcn_wmma_f32_16x16x32_f16(false, (A), false, (B), (short)0, (C), false, false)

__device__ __forceinline__ void stage_chunk(const float* __restrict__ xg, const float* __restrict__ yg,
                                            float* xtile, float* ytile, int chunk, int buf, int tid) {
    int t0 = chunk * CH;
    for (int idx = tid; idx < NB * CH; idx += 256) {
        int b = idx >> 5, tt = idx & (CH - 1);
        ASYNC_CP_B32(xg + b * NL + t0 + tt, xtile + buf * (NB * CH) + idx);
        ASYNC_CP_B32(yg + b * NL + t0 + tt, ytile + buf * (NB * CH) + idx);
    }
}

__global__ __launch_bounds__(256, 1)
void disc_scan_kernel(const float* __restrict__ xg,  const float* __restrict__ yg,
                      const float* __restrict__ s0g, const float* __restrict__ e0g,
                      const float* __restrict__ g0g, const float* __restrict__ Wcg,
                      const float* __restrict__ Uxg, const float* __restrict__ bcg,
                      const float* __restrict__ mug, const float* __restrict__ Wx1g,
                      const float* __restrict__ Wh1g,const float* __restrict__ b1g,
                      const float* __restrict__ Wx2g,const float* __restrict__ Wh2g,
                      const float* __restrict__ b2g, const float* __restrict__ Wog,
                      const float* __restrict__ bog, const float* __restrict__ Wling,
                      const float* __restrict__ bling,const float* __restrict__ Wd1g,
                      const float* __restrict__ bd1g,const float* __restrict__ Wd2g,
                      const float* __restrict__ bd2g,float* __restrict__ out)
{
    extern __shared__ __align__(128) unsigned char smem[];
    _Float16* WcT    = (_Float16*)(smem + OFF_WCT);
    _Float16* WxT1   = (_Float16*)(smem + OFF_WXT1);
    _Float16* WhT1   = (_Float16*)(smem + OFF_WHT1);
    _Float16* WxT2   = (_Float16*)(smem + OFF_WXT2);
    _Float16* WhT2   = (_Float16*)(smem + OFF_WHT2);
    _Float16* WoT    = (_Float16*)(smem + OFF_WOT);
    _Float16* WlT    = (_Float16*)(smem + OFF_WLT);
    _Float16* Wd1T   = (_Float16*)(smem + OFF_WD1T);
    _Float16* Wd2T   = (_Float16*)(smem + OFF_WD2T);
    _Float16* inp16  = (_Float16*)(smem + OFF_INP16);
    _Float16* h1_16  = (_Float16*)(smem + OFF_H1_16);
    _Float16* h2_16  = (_Float16*)(smem + OFF_H2_16);
    _Float16* st16   = (_Float16*)(smem + OFF_ST16);
    _Float16* tmpo16 = (_Float16*)(smem + OFF_TMPO16);
    _Float16* co16   = (_Float16*)(smem + OFF_CO16);
    _Float16* hid16  = (_Float16*)(smem + OFF_HID16);
    float* state  = (float*)(smem + OFF_STATE);
    float* stnew  = (float*)(smem + OFF_STNEW);
    float* zbuf   = (float*)(smem + OFF_Z);
    float* c1f    = (float*)(smem + OFF_C1F);
    float* c2f    = (float*)(smem + OFF_C2F);
    float* inp32  = (float*)(smem + OFF_INP32);
    float* gateA  = (float*)(smem + OFF_GATEA);
    float* gateB  = (float*)(smem + OFF_GATEB);
    float* gsm    = (float*)(smem + OFF_GSM);
    float* co32   = (float*)(smem + OFF_CO32);
    float* uxL    = (float*)(smem + OFF_UXL);
    float* bcL    = (float*)(smem + OFF_BCL);
    float* muL    = (float*)(smem + OFF_MUL);
    float* b1L    = (float*)(smem + OFF_B1L);
    float* b2L    = (float*)(smem + OFF_B2L);
    float* boL    = (float*)(smem + OFF_BOL);
    float* blinL  = (float*)(smem + OFF_BLINL);
    float* bd1L   = (float*)(smem + OFF_BD1L);
    float* bd2L   = (float*)(smem + OFF_BD2L);
    float* xs     = (float*)(smem + OFF_XS);
    float* erb    = (float*)(smem + OFF_ERR);
    float* thb    = (float*)(smem + OFF_TH);
    float* xtile  = (float*)(smem + OFF_XT);
    float* ytile  = (float*)(smem + OFF_YT);

    const int tid  = threadIdx.x;
    const int wv   = tid >> 5;
    const int lane = tid & 31;
    const int hi8  = ((lane >> 4) & 1) << 3;
    const int nloc = lane & 15;

    // ---------------- init ----------------
    stage_chunk(xg, yg, xtile, ytile, 0, 0, tid);   // overlap with weight staging below

    for (int idx = tid; idx < NC * NS * NS; idx += 256) {       // WcT[c][d][s] = Wc[c][s][d]
        int c = idx >> 10, d = (idx >> 5) & 31, s = idx & 31;
        WcT[idx] = (_Float16)Wcg[(c << 10) + (s << 5) + d];
    }
    for (int idx = tid; idx < 128 * 32; idx += 256) {           // LSTM weights [n][k]
        int n = idx >> 5, k = idx & 31;
        WxT1[idx] = (k < NTC + 1) ? (_Float16)Wx1g[k * 128 + n] : (_Float16)0.0f;
        WhT1[idx] = (_Float16)Wh1g[k * 128 + n];
        WxT2[idx] = (_Float16)Wx2g[k * 128 + n];
        WhT2[idx] = (_Float16)Wh2g[k * 128 + n];
    }
    for (int idx = tid; idx < 32 * 32; idx += 256) {            // WoT / WlinT
        int n = idx >> 5, k = idx & 31;
        WoT[idx] = (n < NTC + 1) ? (_Float16)Wog[k * (NTC + 1) + n] : (_Float16)0.0f;
        WlT[idx] = (k < NTC + 1) ? (_Float16)Wling[k * NH + n]      : (_Float16)0.0f;
    }
    for (int idx = tid; idx < 64 * 32; idx += 256) {            // Wd1T
        int n = idx >> 5, k = idx & 31;
        Wd1T[idx] = (k < NH - 1) ? (_Float16)Wd1g[k * 64 + n] : (_Float16)0.0f;
    }
    for (int idx = tid; idx < 64 * 64; idx += 256)              // Wd2T
        Wd2T[idx] = (_Float16)Wd2g[(idx & 63) * NC + (idx >> 6)];
    for (int idx = tid; idx < NC * NS; idx += 256) { uxL[idx] = Uxg[idx]; bcL[idx] = bcg[idx]; }
    for (int idx = tid; idx < NTC * NS; idx += 256) muL[idx] = mug[idx];
    for (int idx = tid; idx < 128; idx += 256) { b1L[idx] = b1g[idx]; b2L[idx] = b2g[idx]; }
    if (tid < NTC + 1) boL[tid] = bog[tid];
    if (tid < NH)      blinL[tid] = bling[tid];
    if (tid < 64)      { bd1L[tid] = bd1g[tid]; bd2L[tid] = bd2g[tid]; }
    for (int idx = tid; idx < NB * NS; idx += 256) {
        float v = s0g[idx & 31];
        state[idx] = v; st16[idx] = (_Float16)v;
        c1f[idx] = 0.f; c2f[idx] = 0.f;
        h1_16[idx] = (_Float16)0.f; h2_16[idx] = (_Float16)0.f;
        inp32[idx] = 0.f; inp16[idx] = (_Float16)0.f;
        tmpo16[idx] = (_Float16)0.f; co16[idx] = (_Float16)0.f;
    }
    for (int idx = tid; idx < NB * NC; idx += 256) { gateA[idx] = g0g[idx]; hid16[idx] = (_Float16)0.f; }
    if (tid < NB) erb[tid] = e0g[tid];
    ASYNC_WAIT();
    __syncthreads();

    for (int t = 0; t < NL; ++t) {
        const int chunk = t / CH, pos = t & (CH - 1), buf = chunk & 1;
        float* gcur = (t & 1) ? gateB : gateA;
        float* gnxt = (t & 1) ? gateA : gateB;

        // kick off next chunk's x/y staging (overlaps the next 32 steps)
        if (pos == 0 && chunk + 1 < NL / CH)
            stage_chunk(xg, yg, xtile, ytile, chunk + 1, buf ^ 1, tid);

        // ---- A0: x_t, RBF encoding, error column ----
        if (tid < NB) xs[tid] = xtile[buf * (NB * CH) + tid * CH + pos];
        for (int idx = tid; idx < NB * NTC; idx += 256) {
            int b = idx / NTC, j = idx % NTC;
            const float* sr = state + b * NS;
            const float* mr = muL + j * NS;
            float acc = 0.f;
            #pragma unroll
            for (int s = 0; s < NS; ++s) { float d = sr[s] - mr[s]; acc += d * d; }
            inp32[b * 32 + j] = __expf(-GAMMA_C * acc);
        }
        if (tid < NB) inp32[tid * 32 + NTC] = erb[tid];
        __syncthreads();
        for (int idx = tid; idx < NB * (NTC + 1); idx += 256) {
            int b = idx / (NTC + 1), j = idx % (NTC + 1);
            inp16[b * 32 + j] = (_Float16)inp32[b * 32 + j];
        }
        __syncthreads();

        // ---- A1: z = inp@Wx1 + h1@Wh1 + b1 (WMMA) ----
        #pragma unroll
        for (int p = 0; p < 2; ++p) {
            int tI = wv * 2 + p, mt = tI >> 3, nt = tI & 7;
            v16h a0 = ldsFrag16s(inp16, mt * 16, 32, lane);
            v16h b0 = ldsFrag16s(WxT1 + nt * 512, 0, 32, lane);
            v8f acc = {};
            acc = WMMA16(a0, b0, acc);
            v16h a1 = ldsFrag16s(h1_16, mt * 16, 32, lane);
            v16h b1v = ldsFrag16s(WhT1 + nt * 512, 0, 32, lane);
            acc = WMMA16(a1, b1v, acc);
            int n = nt * 16 + nloc;
            float bias = b1L[n];
            #pragma unroll
            for (int r = 0; r < 8; ++r) zbuf[(mt * 16 + hi8 + r) * 128 + n] = acc[r] + bias;
        }
        __syncthreads();
        // ---- A2: LSTM1 elementwise ----
        for (int idx = tid; idx < NB * NH; idx += 256) {
            int b = idx >> 5, hh = idx & 31;
            const float* zr = zbuf + b * 128;
            float iv = sigm(zr[hh]), fv = sigm(zr[32 + hh]);
            float gv = TANHF(zr[64 + hh]), ov = sigm(zr[96 + hh]);
            float cn = fv * c1f[idx] + iv * gv;
            float hn = ov * TANHF(cn);
            c1f[idx] = cn; h1_16[idx] = (_Float16)hn;
        }
        __syncthreads();

        // ---- A3: z = h1@Wx2 + h2@Wh2 + b2 ----
        #pragma unroll
        for (int p = 0; p < 2; ++p) {
            int tI = wv * 2 + p, mt = tI >> 3, nt = tI & 7;
            v16h a0 = ldsFrag16s(h1_16, mt * 16, 32, lane);
            v16h b0 = ldsFrag16s(WxT2 + nt * 512, 0, 32, lane);
            v8f acc = {};
            acc = WMMA16(a0, b0, acc);
            v16h a1 = ldsFrag16s(h2_16, mt * 16, 32, lane);
            v16h b1v = ldsFrag16s(WhT2 + nt * 512, 0, 32, lane);
            acc = WMMA16(a1, b1v, acc);
            int n = nt * 16 + nloc;
            float bias = b2L[n];
            #pragma unroll
            for (int r = 0; r < 8; ++r) zbuf[(mt * 16 + hi8 + r) * 128 + n] = acc[r] + bias;
        }
        __syncthreads();
        // ---- A4: LSTM2 elementwise ----
        for (int idx = tid; idx < NB * NH; idx += 256) {
            int b = idx >> 5, hh = idx & 31;
            const float* zr = zbuf + b * 128;
            float iv = sigm(zr[hh]), fv = sigm(zr[32 + hh]);
            float gv = TANHF(zr[64 + hh]), ov = sigm(zr[96 + hh]);
            float cn = fv * c2f[idx] + iv * gv;
            float hn = ov * TANHF(cn);
            c2f[idx] = cn; h2_16[idx] = (_Float16)hn;
        }
        __syncthreads();

        // ---- A5: tmpo = h2@Wo + bo (WMMA, N valid < 18) ----
        if (wv < 4) {
            int mt = wv >> 1, nt = wv & 1;
            v16h a = ldsFrag16s(h2_16, mt * 16, 32, lane);
            v16h b = ldsFrag16s(WoT + nt * 512, 0, 32, lane);
            v8f acc = {};
            acc = WMMA16(a, b, acc);
            int n = nt * 16 + nloc;
            if (n < NTC + 1) {
                float bias = boL[n];
                #pragma unroll
                for (int r = 0; r < 8; ++r)
                    tmpo16[(mt * 16 + hi8 + r) * 32 + n] = (_Float16)(acc[r] + bias);
            }
        }
        __syncthreads();
        // ---- A6: co = tmpo@Wlin + blin (WMMA) ----
        if (wv < 4) {
            int mt = wv >> 1, nt = wv & 1;
            v16h a = ldsFrag16s(tmpo16, mt * 16, 32, lane);
            v16h b = ldsFrag16s(WlT + nt * 512, 0, 32, lane);
            v8f acc = {};
            acc = WMMA16(a, b, acc);
            int n = nt * 16 + nloc;
            float bias = blinL[n];
            #pragma unroll
            for (int r = 0; r < 8; ++r) {
                int m = mt * 16 + hi8 + r;
                float v = acc[r] + bias;
                co32[m * 32 + n] = v;
                co16[m * 32 + n] = (_Float16)v;
            }
        }
        __syncthreads();
        // ---- A7: hid = relu(co[:, :31]@Wd1 + bd1) (WMMA) ----
        {
            int mt = wv >> 2, nt = wv & 3;
            v16h a = ldsFrag16s(co16, mt * 16, 32, lane);
            v16h b = ldsFrag16s(Wd1T + nt * 512, 0, 32, lane);
            v8f acc = {};
            acc = WMMA16(a, b, acc);
            int n = nt * 16 + nloc;
            float bias = bd1L[n];
            #pragma unroll
            for (int r = 0; r < 8; ++r)
                hid16[(mt * 16 + hi8 + r) * 64 + n] = (_Float16)fmaxf(acc[r] + bias, 0.f);
        }
        __syncthreads();
        // ---- A8: gsm = hid@Wd2 + bd2 (WMMA, 2 K-steps); theta ----
        {
            int mt = wv >> 2, nt = wv & 3;
            v8f acc = {};
            #pragma unroll
            for (int kk = 0; kk < 2; ++kk) {
                v16h a = ldsFrag16s(hid16 + kk * 32, mt * 16, 64, lane);
                v16h b = ldsFrag16s(Wd2T + nt * 1024 + kk * 32, 0, 64, lane);
                acc = WMMA16(a, b, acc);
            }
            int n = nt * 16 + nloc;
            float bias = bd2L[n];
            #pragma unroll
            for (int r = 0; r < 8; ++r) gsm[(mt * 16 + hi8 + r) * 64 + n] = acc[r] + bias;
        }
        if (tid < NB) thb[tid] = sigm(co32[tid * 32 + (NH - 1)]);
        __syncthreads();
        // ---- A9: softmax rows over C ----
        if (tid < NB) {
            float* row = gsm + tid * NC;
            float mx = row[0];
            for (int j = 1; j < NC; ++j) mx = fmaxf(mx, row[j]);
            float ssum = 0.f;
            for (int j = 0; j < NC; ++j) { float e = __expf(row[j] - mx); row[j] = e; ssum += e; }
            float inv = RCPF(ssum);
            for (int j = 0; j < NC; ++j) row[j] *= inv;
        }
        __syncthreads();
        // ---- A10: gate update; zero state accumulator ----
        for (int idx = tid; idx < NB * NC; idx += 256) {
            int b = idx >> 6;
            float th = thb[b];
            gnxt[idx] = gsm[idx] * th + gcur[idx] * (1.f - th);
        }
        for (int idx = tid; idx < NB * NS; idx += 256) stnew[idx] = 0.f;
        __syncthreads();

        // ---- B: fused einsum — tanh(state@Wc[c] + x_t*Ux[c] + bc[c]) contracted with gate ----
        {
            v16h aFrag0 = ldsFrag16s(st16, 0, 32, lane);
            v16h aFrag1 = ldsFrag16s(st16, 16, 32, lane);
            float accS[4][8];
            #pragma unroll
            for (int q = 0; q < 4; ++q)
                #pragma unroll
                for (int r = 0; r < 8; ++r) accS[q][r] = 0.f;
            for (int ci = 0; ci < 8; ++ci) {
                int c = wv + ci * 8;                 // each wave owns 8 of C=64
                #pragma unroll
                for (int mt = 0; mt < 2; ++mt) {
                    v16h a = mt ? aFrag1 : aFrag0;
                    #pragma unroll
                    for (int nt = 0; nt < 2; ++nt) {
                        v16h b = ldsFrag16s(WcT + c * 1024 + nt * 512, 0, 32, lane);
                        v8f acc = {};
                        acc = WMMA16(a, b, acc);
                        int n = nt * 16 + nloc;
                        float uxv = uxL[c * NS + n];
                        float bcv = bcL[c * NS + n];
                        #pragma unroll
                        for (int r = 0; r < 8; ++r) {
                            int m = mt * 16 + hi8 + r;
                            float v = TANHF(acc[r] + xs[m] * uxv + bcv);
                            accS[mt * 2 + nt][r] += gnxt[m * NC + c] * v;
                        }
                    }
                }
            }
            #pragma unroll
            for (int mt = 0; mt < 2; ++mt)
                #pragma unroll
                for (int nt = 0; nt < 2; ++nt) {
                    int n = nt * 16 + nloc;
                    #pragma unroll
                    for (int r = 0; r < 8; ++r) {
                        int m = mt * 16 + hi8 + r;
                        __hip_atomic_fetch_add(&stnew[m * NS + n], accS[mt * 2 + nt][r],
                                               __ATOMIC_RELAXED, __HIP_MEMORY_SCOPE_WORKGROUP);
                    }
                }
        }
        __syncthreads();

        // ---- C: prediction, error, commit state ----
        if (tid < NB) {
            float p = stnew[tid * NS + (NS - 1)];
            float pred = fminf(fmaxf(p, 0.f), 1.f);
            erb[tid] = pred - ytile[buf * (NB * CH) + tid * CH + pos];
            out[tid * NL + t] = pred;
        }
        for (int idx = tid; idx < NB * NS; idx += 256) {
            float s = stnew[idx];
            state[idx] = s; st16[idx] = (_Float16)s;
        }
        if (pos == CH - 1) ASYNC_WAIT();   // next chunk's staging must be in LDS
        __syncthreads();
    }
}

extern "C" void kernel_launch(void* const* d_in, const int* in_sizes, int n_in,
                              void* d_out, int out_size, void* d_ws, size_t ws_size,
                              hipStream_t stream) {
    (void)in_sizes; (void)n_in; (void)d_ws; (void)ws_size; (void)out_size;
    const float* x    = (const float*)d_in[0];
    const float* y    = (const float*)d_in[1];
    const float* s0   = (const float*)d_in[2];
    const float* err0 = (const float*)d_in[3];
    const float* g0   = (const float*)d_in[4];
    const float* Wc   = (const float*)d_in[5];
    const float* Ux   = (const float*)d_in[6];
    const float* bc   = (const float*)d_in[7];
    const float* mu   = (const float*)d_in[8];
    const float* Wx1  = (const float*)d_in[9];
    const float* Wh1  = (const float*)d_in[10];
    const float* b1   = (const float*)d_in[11];
    const float* Wx2  = (const float*)d_in[12];
    const float* Wh2  = (const float*)d_in[13];
    const float* b2   = (const float*)d_in[14];
    const float* Wo   = (const float*)d_in[15];
    const float* bo   = (const float*)d_in[16];
    const float* Wlin = (const float*)d_in[17];
    const float* blin = (const float*)d_in[18];
    const float* Wd1  = (const float*)d_in[19];
    const float* bd1  = (const float*)d_in[20];
    const float* Wd2  = (const float*)d_in[21];
    const float* bd2  = (const float*)d_in[22];
    float* out = (float*)d_out;

    disc_scan_kernel<<<dim3(1), dim3(256), SMEM_TOTAL, stream>>>(
        x, y, s0, err0, g0, Wc, Ux, bc, mu, Wx1, Wh1, b1, Wx2, Wh2, b2,
        Wo, bo, Wlin, blin, Wd1, bd1, Wd2, bd2, out);
}
